// PointerNetwork_19018115186940
// MI455X (gfx1250) — matrix operand order repaired and checked by reading, not compiled, not run on hardware
//
#include <hip/hip_runtime.h>

typedef _Float16 half_t;
typedef __attribute__((ext_vector_type(16))) _Float16 v16h;
typedef __attribute__((ext_vector_type(8)))  _Float16 v8h;
typedef __attribute__((ext_vector_type(8)))  float    v8f;

#define B_DIM 1024
#define S_DIM 128
#define E_DIM 256
#define H_DIM 256
#define MASK_FILL -100000.0f

// ---------------------------------------------------------------------------
// Async global->LDS staging (CDNA5 path when available, safe fallback else)
// ---------------------------------------------------------------------------
typedef __attribute__((address_space(1))) int gas_int;
typedef __attribute__((address_space(3))) int las_int;

__device__ __forceinline__ void stage_f32(const float* __restrict__ g, float* l)
{
#if __has_builtin(__builtin_amdgcn_global_load_async_to_lds_b32)
  __builtin_amdgcn_global_load_async_to_lds_b32(
      (gas_int*)g, (las_int*)l, 0, 0);
#else
  *l = *g;
#endif
}

__device__ __forceinline__ void stage_wait()
{
#if __has_builtin(__builtin_amdgcn_global_load_async_to_lds_b32)
#if __has_builtin(__builtin_amdgcn_s_wait_asynccnt)
  __builtin_amdgcn_s_wait_asynccnt(0);
#else
  asm volatile("s_wait_asynccnt 0x0" ::: "memory");
#endif
#endif
}

// ---------------------------------------------------------------------------
// Dual-input WMMA GEMM:
//   C[M,N] = A1[M,K1] @ W1[N,K1]^T  (+ A2[M,K2] @ W2[N,K2]^T if A2 != null)
// f16 inputs, f32 accumulate via v_wmma_f32_16x16x32_f16; one wave = one
// 16x16 C tile. Optional per-column f32 bias; output f32 or f16.
// ---------------------------------------------------------------------------
__global__ __launch_bounds__(128) void gemm2_wmma_kernel(
    const half_t* __restrict__ A1, long lda1, const half_t* __restrict__ W1,
    const half_t* __restrict__ A2, long lda2, const half_t* __restrict__ W2,
    const float* __restrict__ bias,
    float* __restrict__ Cf, half_t* __restrict__ Ch, long ldc,
    int M, int N, int K1, int K2)
{
  int lane = threadIdx.x & 31;
  int wave = threadIdx.x >> 5;
  int nTiles = N >> 4;
  long totalTiles = (long)(M >> 4) * nTiles;
  long tile = (long)blockIdx.x * 4 + wave;
  if (tile >= totalTiles) return;          // uniform per wave: EXEC stays full
  int m0 = (int)(tile / nTiles) << 4;
  int n0 = (int)(tile % nTiles) << 4;

  // A fragment (16x32 f16): lanes 0-15 row m0+lane, K 0..15;
  // lanes 16-31 row m0+lane-16, K 16..31 (matches B).
  int rowOff = lane & 15;
  int kOff = (lane >> 4) << 4;             // 0 or 16
  // B fragment (32x16 f16): lane col n0+(lane&15); lanes 0-15 K 0..15,
  // lanes 16-31 K 16..31, K-major in the 16 halves.
  int nB = n0 + (lane & 15);
  int kBase = (lane >> 4) << 4;            // 0 or 16

  v8f acc = {};
  {
    const half_t* aRow = A1 + (long)(m0 + rowOff) * lda1 + kOff;
    const half_t* bRow = W1 + (long)nB * K1 + kBase;
    for (int k0 = 0; k0 < K1; k0 += 32) {
      v8h alo = *(const v8h*)(aRow + k0);
      v8h ahi = *(const v8h*)(aRow + k0 + 8);
      v16h a;
#pragma unroll
      for (int i = 0; i < 8; ++i) { a[i] = alo[i]; a[i + 8] = ahi[i]; }
      v16h b = *(const v16h*)(bRow + k0);
      acc = __builtin_amdgcn_wmma_f32_16x16x32_f16(
          false, a, false, b, (short)0, acc, false, false);
    }
  }
  if (A2) {
    const half_t* aRow = A2 + (long)(m0 + rowOff) * lda2 + kOff;
    const half_t* bRow = W2 + (long)nB * K2 + kBase;
    for (int k0 = 0; k0 < K2; k0 += 32) {
      v8h alo = *(const v8h*)(aRow + k0);
      v8h ahi = *(const v8h*)(aRow + k0 + 8);
      v16h a;
#pragma unroll
      for (int i = 0; i < 8; ++i) { a[i] = alo[i]; a[i + 8] = ahi[i]; }
      v16h b = *(const v16h*)(bRow + k0);
      acc = __builtin_amdgcn_wmma_f32_16x16x32_f16(
          false, a, false, b, (short)0, acc, false, false);
    }
  }

  int col = n0 + (lane & 15);
  float bv = bias ? bias[col] : 0.0f;
  int rBase = m0 + ((lane >> 4) << 3);     // VGPR r -> row rBase+r
#pragma unroll
  for (int r = 0; r < 8; ++r) {
    long idx = (long)(rBase + r) * ldc + col;
    float val = acc[r] + bv;
    if (Ch) Ch[idx] = (half_t)val;
    else    Cf[idx] = val;
  }
}

// ---------------------------------------------------------------------------
// embedded = x @ emb_W  (K=2: scalar math, store f16)
// ---------------------------------------------------------------------------
__global__ void embed_kernel(const float* __restrict__ x,
                             const float* __restrict__ embW,
                             half_t* __restrict__ emb16, long n, int E)
{
  long idx = (long)blockIdx.x * blockDim.x + threadIdx.x;
  if (idx >= n) return;
  long bs = idx / E; int e = (int)(idx % E);
  float v = x[bs * 2] * embW[e] + x[bs * 2 + 1] * embW[E + e];
  emb16[idx] = (half_t)v;
}

__global__ void f32_to_f16_kernel(const float* __restrict__ src,
                                  half_t* __restrict__ dst, long n)
{
  long idx = (long)blockIdx.x * blockDim.x + threadIdx.x;
  if (idx < n) dst[idx] = (half_t)src[idx];
}

__global__ void init_state_kernel(float* h, float* c, half_t* h16,
                                  unsigned char* mask, half_t* dec_in16,
                                  const float* __restrict__ dec_start)
{
  long idx = (long)blockIdx.x * blockDim.x + threadIdx.x;
  if (idx < (long)B_DIM * H_DIM) {
    h[idx] = 0.0f; c[idx] = 0.0f; h16[idx] = (half_t)0.0f;
  }
  if (idx < (long)B_DIM * S_DIM) mask[idx] = 0;
  if (idx < (long)B_DIM * E_DIM) {
    int e = (int)(idx % E_DIM);
    dec_in16[idx] = (half_t)dec_start[e];
  }
}

// ---------------------------------------------------------------------------
// LSTM cell elementwise: gates hold x@Wih^T + h@Whh^T (no bias), order i,f,g,o
// ---------------------------------------------------------------------------
__global__ __launch_bounds__(256) void lstm_cell_kernel(
    const float* __restrict__ gates,
    const float* __restrict__ b_ih, const float* __restrict__ b_hh,
    float* __restrict__ c, float* __restrict__ h, half_t* __restrict__ h16,
    half_t* __restrict__ encOutSlice, long encStride)
{
  long idx = (long)blockIdx.x * blockDim.x + threadIdx.x;
  if (idx >= (long)B_DIM * H_DIM) return;
  int b = (int)(idx / H_DIM), j = (int)(idx % H_DIM);
  const float* g = gates + (long)b * 4 * H_DIM;
  float iv = g[j]             + b_ih[j]             + b_hh[j];
  float fv = g[H_DIM + j]     + b_ih[H_DIM + j]     + b_hh[H_DIM + j];
  float gv = g[2 * H_DIM + j] + b_ih[2 * H_DIM + j] + b_hh[2 * H_DIM + j];
  float ov = g[3 * H_DIM + j] + b_ih[3 * H_DIM + j] + b_hh[3 * H_DIM + j];
  iv = 1.0f / (1.0f + __expf(-iv));
  fv = 1.0f / (1.0f + __expf(-fv));
  gv = tanhf(gv);
  ov = 1.0f / (1.0f + __expf(-ov));
  float cn = fv * c[idx] + iv * gv;
  float hn = ov * tanhf(cn);
  c[idx] = cn; h[idx] = hn; h16[idx] = (half_t)hn;
  if (encOutSlice) encOutSlice[(long)b * encStride + j] = (half_t)hn;
}

// ---------------------------------------------------------------------------
// Fused glimpse head: one block (256 thr) per batch row.
//   logits_s = C*tanh(v . tanh(q_b + keys_bs) + vb)  (masked)
//   a = softmax(logits); query = sum_s a_s * keys_bs  -> f16 for next GEMM
// ---------------------------------------------------------------------------
__global__ __launch_bounds__(256) void attn_glimpse_kernel(
    const float* __restrict__ q, const half_t* __restrict__ keys,
    const float* __restrict__ v_w, const float* __restrict__ v_b,
    const unsigned char* __restrict__ mask,
    half_t* __restrict__ query16, float Cmul)
{
  __shared__ float qs[H_DIM];
  __shared__ float vs[H_DIM];
  __shared__ float lsh[S_DIM];
  __shared__ float prob[S_DIM];
  __shared__ float red[S_DIM];
  int b = blockIdx.x, tid = threadIdx.x;
  int lane = tid & 31, wave = tid >> 5;

  stage_f32(q + (long)b * H_DIM + tid, qs + tid);
  vs[tid] = v_w[tid];
  stage_wait();
  __syncthreads();

  const half_t* kb = keys + (long)b * S_DIM * H_DIM;
  for (int i = 0; i < 16; ++i) {
    int s = wave * 16 + i;
    const half_t* kr = kb + (long)s * H_DIM;
    __builtin_prefetch(kr + H_DIM, 0, 1);
    float sum = 0.0f;
    for (int j = lane; j < H_DIM; j += 32)
      sum += tanhf(qs[j] + (float)kr[j]) * vs[j];
#pragma unroll
    for (int off = 16; off >= 1; off >>= 1) sum += __shfl_xor(sum, off, 32);
    if (lane == 0) {
      float lg = Cmul * tanhf(sum + v_b[0]);
      lsh[s] = mask[(long)b * S_DIM + s] ? MASK_FILL : lg;
    }
  }
  __syncthreads();
  if (tid < S_DIM) red[tid] = lsh[tid];
  __syncthreads();
  for (int off = 64; off >= 1; off >>= 1) {
    if (tid < off) red[tid] = fmaxf(red[tid], red[tid + off]);
    __syncthreads();
  }
  float mx = red[0]; __syncthreads();
  if (tid < S_DIM) { prob[tid] = __expf(lsh[tid] - mx); red[tid] = prob[tid]; }
  __syncthreads();
  for (int off = 64; off >= 1; off >>= 1) {
    if (tid < off) red[tid] += red[tid + off];
    __syncthreads();
  }
  float inv = 1.0f / red[0];
  if (tid < S_DIM) prob[tid] *= inv;
  __syncthreads();
  float acc = 0.0f;
  for (int ss = 0; ss < S_DIM; ++ss)
    acc += prob[ss] * (float)kb[(long)ss * H_DIM + tid];
  query16[(long)b * H_DIM + tid] = (half_t)acc;
}

// ---------------------------------------------------------------------------
// Fused pointer head: logits -> log-softmax -> Gumbel-max sample (hash RNG)
// -> mask update, lp/idx output, gather next decoder input.
// ---------------------------------------------------------------------------
__device__ __forceinline__ unsigned int hash3(unsigned int a, unsigned int b,
                                              unsigned int c) {
  unsigned int h = a * 0x9E3779B1u ^ b * 0x85EBCA77u ^ c * 0xC2B2AE3Du;
  h ^= h >> 16; h *= 0x7FEB352Du; h ^= h >> 15; h *= 0x846CA68Bu; h ^= h >> 16;
  return h;
}

__global__ __launch_bounds__(256) void attn_sample_kernel(
    const float* __restrict__ q, const half_t* __restrict__ keys,
    const float* __restrict__ v_w, const float* __restrict__ v_b,
    unsigned char* __restrict__ mask,
    const half_t* __restrict__ emb16, half_t* __restrict__ dec_in16,
    float* __restrict__ lps, int* __restrict__ idxs, int t, float Cmul)
{
  __shared__ float qs[H_DIM];
  __shared__ float vs[H_DIM];
  __shared__ float lsh[S_DIM];
  __shared__ float slp[S_DIM];
  __shared__ float red[S_DIM];
  __shared__ int redi[S_DIM];
  __shared__ int chosen_s;
  int b = blockIdx.x, tid = threadIdx.x;
  int lane = tid & 31, wave = tid >> 5;

  stage_f32(q + (long)b * H_DIM + tid, qs + tid);
  vs[tid] = v_w[tid];
  stage_wait();
  __syncthreads();

  const half_t* kb = keys + (long)b * S_DIM * H_DIM;
  for (int i = 0; i < 16; ++i) {
    int s = wave * 16 + i;
    const half_t* kr = kb + (long)s * H_DIM;
    __builtin_prefetch(kr + H_DIM, 0, 1);
    float sum = 0.0f;
    for (int j = lane; j < H_DIM; j += 32)
      sum += tanhf(qs[j] + (float)kr[j]) * vs[j];
#pragma unroll
    for (int off = 16; off >= 1; off >>= 1) sum += __shfl_xor(sum, off, 32);
    if (lane == 0) {
      float lg = Cmul * tanhf(sum + v_b[0]);
      lsh[s] = mask[(long)b * S_DIM + s] ? MASK_FILL : lg;
    }
  }
  __syncthreads();
  if (tid < S_DIM) red[tid] = lsh[tid];
  __syncthreads();
  for (int off = 64; off >= 1; off >>= 1) {
    if (tid < off) red[tid] = fmaxf(red[tid], red[tid + off]);
    __syncthreads();
  }
  float mx = red[0]; __syncthreads();
  if (tid < S_DIM) red[tid] = __expf(lsh[tid] - mx);
  __syncthreads();
  for (int off = 64; off >= 1; off >>= 1) {
    if (tid < off) red[tid] += red[tid + off];
    __syncthreads();
  }
  float logZ = mx + __logf(red[0]); __syncthreads();
  if (tid < S_DIM) {
    slp[tid] = lsh[tid] - logZ;
    unsigned int hsh = hash3((unsigned)t + 1u, (unsigned)b, (unsigned)tid);
    float u = ((float)hsh + 0.5f) * 2.3283064365386963e-10f; // (0,1)
    float g = -__logf(-__logf(u));
    red[tid] = lsh[tid] + g;
    redi[tid] = tid;
  }
  __syncthreads();
  for (int off = 64; off >= 1; off >>= 1) {
    if (tid < off && red[tid + off] > red[tid]) {
      red[tid] = red[tid + off]; redi[tid] = redi[tid + off];
    }
    __syncthreads();
  }
  if (tid == 0) {
    int ch = redi[0];
    chosen_s = ch;
    mask[(long)b * S_DIM + ch] = 1;
    lps[(long)b * S_DIM + t] = slp[ch];
    idxs[(long)b * S_DIM + t] = ch;
  }
  __syncthreads();
  int ch = chosen_s;
  dec_in16[(long)b * E_DIM + tid] =
      emb16[((long)b * S_DIM + ch) * E_DIM + tid];
}

// ---------------------------------------------------------------------------
// Host orchestration
// ---------------------------------------------------------------------------
static inline size_t alignUp(size_t x) { return (x + 255) & ~(size_t)255; }

static inline void launch_gemm2(const half_t* A1, long lda1, const half_t* W1,
                                const half_t* A2, long lda2, const half_t* W2,
                                const float* bias, float* Cf, half_t* Ch,
                                long ldc, int M, int N, int K1, int K2,
                                hipStream_t stream) {
  long tiles = (long)(M / 16) * (N / 16);
  int blocks = (int)((tiles + 3) / 4);
  gemm2_wmma_kernel<<<blocks, 128, 0, stream>>>(A1, lda1, W1, A2, lda2, W2,
                                                bias, Cf, Ch, ldc, M, N, K1, K2);
}

extern "C" void kernel_launch(void* const* d_in, const int* in_sizes, int n_in,
                              void* d_out, int out_size, void* d_ws, size_t ws_size,
                              hipStream_t stream) {
  const float* x         = (const float*)d_in[0];
  const float* emb_W     = (const float*)d_in[1];
  const float* enc_W_ih  = (const float*)d_in[2];
  const float* enc_W_hh  = (const float*)d_in[3];
  const float* enc_b_ih  = (const float*)d_in[4];
  const float* enc_b_hh  = (const float*)d_in[5];
  const float* dec_W_ih  = (const float*)d_in[6];
  const float* dec_W_hh  = (const float*)d_in[7];
  const float* dec_b_ih  = (const float*)d_in[8];
  const float* dec_b_hh  = (const float*)d_in[9];
  const float* g_Wq_w    = (const float*)d_in[10];
  const float* g_Wq_b    = (const float*)d_in[11];
  const float* g_Wk_w    = (const float*)d_in[12];
  const float* g_Wk_b    = (const float*)d_in[13];
  const float* g_v_w     = (const float*)d_in[14];
  const float* g_v_b     = (const float*)d_in[15];
  const float* p_Wq_w    = (const float*)d_in[16];
  const float* p_Wq_b    = (const float*)d_in[17];
  const float* p_Wk_w    = (const float*)d_in[18];
  const float* p_Wk_b    = (const float*)d_in[19];
  const float* p_v_w     = (const float*)d_in[20];
  const float* p_v_b     = (const float*)d_in[21];
  const float* dec_start = (const float*)d_in[22];

  const int B = B_DIM, S = S_DIM, E = E_DIM, H = H_DIM;
  const long BS = (long)B * S;

  // workspace layout
  char* w = (char*)d_ws;
  size_t off = 0;
  auto alloc = [&](size_t bytes) -> char* {
    char* p = w + off; off = alignUp(off + bytes); return p;
  };
  half_t* emb16     = (half_t*)alloc(BS * E * 2);
  half_t* encOut16  = (half_t*)alloc(BS * H * 2);
  half_t* gkeys16   = (half_t*)alloc(BS * H * 2);
  half_t* pkeys16   = (half_t*)alloc(BS * H * 2);
  half_t* wEncIh16  = (half_t*)alloc((size_t)4 * H * E * 2);
  half_t* wEncHh16  = (half_t*)alloc((size_t)4 * H * H * 2);
  half_t* wDecIh16  = (half_t*)alloc((size_t)4 * H * E * 2);
  half_t* wDecHh16  = (half_t*)alloc((size_t)4 * H * H * 2);
  half_t* wGWq16    = (half_t*)alloc((size_t)H * H * 2);
  half_t* wGWk16    = (half_t*)alloc((size_t)H * H * 2);
  half_t* wPWq16    = (half_t*)alloc((size_t)H * H * 2);
  half_t* wPWk16    = (half_t*)alloc((size_t)H * H * 2);
  float*  hbuf      = (float*)alloc((size_t)B * H * 4);
  float*  cbuf      = (float*)alloc((size_t)B * H * 4);
  half_t* h16       = (half_t*)alloc((size_t)B * H * 2);
  float*  gates     = (float*)alloc((size_t)B * 4 * H * 4);
  float*  qbuf      = (float*)alloc((size_t)B * H * 4);
  half_t* query16   = (half_t*)alloc((size_t)B * H * 2);
  half_t* dec_in16  = (half_t*)alloc((size_t)B * E * 2);
  unsigned char* mask = (unsigned char*)alloc(BS);
  if (off > ws_size) return;  // workspace too small; nothing safe to do

  float* lps = (float*)d_out;
  int* idxs = (int*)d_out + BS;

  // ---- prep: embeddings, weight conversion, state init ----
  {
    long n = BS * E;
    embed_kernel<<<(int)((n + 255) / 256), 256, 0, stream>>>(x, emb_W, emb16, n, E);
  }
  auto cvt = [&](const float* src, half_t* dst, long n) {
    f32_to_f16_kernel<<<(int)((n + 255) / 256), 256, 0, stream>>>(src, dst, n);
  };
  cvt(enc_W_ih, wEncIh16, (long)4 * H * E);
  cvt(enc_W_hh, wEncHh16, (long)4 * H * H);
  cvt(dec_W_ih, wDecIh16, (long)4 * H * E);
  cvt(dec_W_hh, wDecHh16, (long)4 * H * H);
  cvt(g_Wq_w, wGWq16, (long)H * H);
  cvt(g_Wk_w, wGWk16, (long)H * H);
  cvt(p_Wq_w, wPWq16, (long)H * H);
  cvt(p_Wk_w, wPWk16, (long)H * H);
  {
    long n = (long)B * H;
    init_state_kernel<<<(int)((n + 255) / 256), 256, 0, stream>>>(
        hbuf, cbuf, h16, mask, dec_in16, dec_start);
  }

  // ---- encoder LSTM: 128 sequential steps (one fused dual GEMM each) ----
  for (int t = 0; t < S; ++t) {
    launch_gemm2(emb16 + (long)t * E, (long)S * E, wEncIh16,
                 h16, H, wEncHh16, nullptr, gates, nullptr, 4 * H,
                 B, 4 * H, E, H, stream);
    long n = (long)B * H;
    lstm_cell_kernel<<<(int)((n + 255) / 256), 256, 0, stream>>>(
        gates, enc_b_ih, enc_b_hh, cbuf, hbuf, h16,
        encOut16 + (long)t * H, (long)S * H);
  }

  // ---- attention keys (f16: both tensors fit in the 192MB L2 together) ----
  launch_gemm2(encOut16, H, wGWk16, nullptr, 0, nullptr, g_Wk_b,
               nullptr, gkeys16, H, (int)BS, H, H, 0, stream);
  launch_gemm2(encOut16, H, wPWk16, nullptr, 0, nullptr, p_Wk_b,
               nullptr, pkeys16, H, (int)BS, H, H, 0, stream);

  // ---- decoder: 128 sequential steps ----
  for (int t = 0; t < S; ++t) {
    launch_gemm2(dec_in16, E, wDecIh16, h16, H, wDecHh16, nullptr,
                 gates, nullptr, 4 * H, B, 4 * H, E, H, stream);
    long n = (long)B * H;
    lstm_cell_kernel<<<(int)((n + 255) / 256), 256, 0, stream>>>(
        gates, dec_b_ih, dec_b_hh, cbuf, hbuf, h16, nullptr, 0);

    // glimpse head
    launch_gemm2(h16, H, wGWq16, nullptr, 0, nullptr, g_Wq_b,
                 qbuf, nullptr, H, B, H, H, 0, stream);
    attn_glimpse_kernel<<<B, 256, 0, stream>>>(
        qbuf, gkeys16, g_v_w, g_v_b, mask, query16, 10.0f);

    // pointer head
    launch_gemm2(query16, H, wPWq16, nullptr, 0, nullptr, p_Wq_b,
                 qbuf, nullptr, H, B, H, H, 0, stream);
    attn_sample_kernel<<<B, 256, 0, stream>>>(
        qbuf, pkeys16, p_v_w, p_v_b, mask, emb16, dec_in16,
        lps, idxs, t, 10.0f);
  }
}
